// SelfAttention_4922032521509
// MI455X (gfx1250) — matrix operand order repaired and checked by reading, not compiled, not run on hardware
//
#include <hip/hip_runtime.h>

// Problem constants (from reference): B=4, S=2048, D=1024
#define D_DIM   1024
#define SEQ     2048
#define BATCH   4
#define MROWS   (BATCH * SEQ)        // 8192 total rows

typedef __attribute__((ext_vector_type(16))) __bf16 bf16x16;
typedef __attribute__((ext_vector_type(8)))  __bf16 bf16x8;
typedef __attribute__((ext_vector_type(8)))  float  f32x8;

// ---------------------------------------------------------------- helpers ---

__device__ __forceinline__ __bf16 f2bf(float f) {
    union { float f; unsigned u; } v; v.f = f;
    unsigned r = v.u + 0x7FFFu + ((v.u >> 16) & 1u);   // round-to-nearest-even
    unsigned short h = (unsigned short)(r >> 16);
    __bf16 o; __builtin_memcpy(&o, &h, 2);
    return o;
}

// A fragment, 16x32 bf16, row-major source.  Layout: lanes 0-15 (hi=0)
// hold K=0..15; lanes 16-31 hold K=16..31; M = lane%16.
// `row` points at element (m, k0).
__device__ __forceinline__ bf16x16 load_a_16x32(const __bf16* row, int hi) {
    bf16x8 lo  = *(const bf16x8*)(row + hi * 16);
    bf16x8 hi8 = *(const bf16x8*)(row + hi * 16 + 8);
    return __builtin_shufflevector(lo, hi8,
        0,1,2,3,4,5,6,7,8,9,10,11,12,13,14,15);
}

// B fragment, 32x16 bf16, from K-major ("transposed") storage: lane holds 16
// consecutive K values of column n=lane%16; lanes 0-15 K=k0..k0+15, lanes
// 16-31 K=k0+16..k0+31.  `col` points at element (k0 + hi*16, n).
__device__ __forceinline__ bf16x16 load_b_32x16(const __bf16* col) {
    bf16x8 b0 = *(const bf16x8*)(col);
    bf16x8 b1 = *(const bf16x8*)(col + 8);
    return __builtin_shufflevector(b0, b1,
        0,1,2,3,4,5,6,7,8,9,10,11,12,13,14,15);
}

__device__ __forceinline__ f32x8 wmma_bf16(bf16x16 a, bf16x16 b, f32x8 c) {
    return __builtin_amdgcn_wmma_f32_16x16x32_bf16(
        false, a, false, b, (short)0, c, false, false);
}

// ------------------------------------------------------------ cvt kernels ---

__global__ __launch_bounds__(256) void cvt_x_kernel(
        const float* __restrict__ X, __bf16* __restrict__ Xb) {
    int i = blockIdx.x * 256 + threadIdx.x;      // MROWS*D_DIM elements
    Xb[i] = f2bf(X[i]);
}

// W stored [in=1024][out=1024]; write transposed Wt[out][in] so WMMA B-frags
// read contiguous K runs.
__global__ __launch_bounds__(256) void cvt_w_kernel(
        const float* __restrict__ Wq, const float* __restrict__ Wk,
        const float* __restrict__ Wv,
        __bf16* __restrict__ WtQ, __bf16* __restrict__ WtK,
        __bf16* __restrict__ WtV) {
    const float* W  = (blockIdx.y == 0) ? Wq  : (blockIdx.y == 1) ? Wk  : Wv;
    __bf16*      Wt = (blockIdx.y == 0) ? WtQ : (blockIdx.y == 1) ? WtK : WtV;
    int i = blockIdx.x * 256 + threadIdx.x;      // D_DIM*D_DIM elements
    int k = i >> 10;       // input (contraction) index
    int n = i & 1023;      // output index
    Wt[n * D_DIM + k] = f2bf(W[i]);
}

// ----------------------------------------------------- Q/K/V projection -----
// Y = Xb @ W + b.  blockIdx.y: 0=Q, 1=K, 2=V.  Each wave owns a 32x32 output
// tile (2x2 WMMA register block): 8 b128 loads feed 4 WMMAs per K-step.
// Q,K stored row-major bf16 [8192][1024].  V stored d-major per batch:
// Vt[b][d][s] so the A.V stage gets contiguous B-fragments.

__global__ __launch_bounds__(256) void proj_kernel(
        const __bf16* __restrict__ Xb,
        const __bf16* __restrict__ WtQ, const float* __restrict__ bq,
        const __bf16* __restrict__ WtK, const float* __restrict__ bk,
        const __bf16* __restrict__ WtV, const float* __restrict__ bv,
        __bf16* __restrict__ Qb, __bf16* __restrict__ Kb,
        __bf16* __restrict__ Vt) {
    const int which = blockIdx.y;
    const __bf16* Wt   = (which == 0) ? WtQ : (which == 1) ? WtK : WtV;
    const float*  bias = (which == 0) ? bq  : (which == 1) ? bk  : bv;

    const int wave = threadIdx.x >> 5;
    const int lane = threadIdx.x & 31;
    const int lo16 = lane & 15;
    const int hi   = lane >> 4;

    const int gtile = blockIdx.x * 8 + wave;     // 8192 tiles (256 x 32)
    const int m0 = (gtile >> 5) * 32;            // 32 n-tiles per row
    const int n0 = (gtile & 31) * 32;

    const __bf16* arow0 = Xb + (size_t)(m0 + lo16) * D_DIM;
    const __bf16* arow1 = arow0 + (size_t)16 * D_DIM;
    const __bf16* bcol0 = Wt + (size_t)(n0 + lo16) * D_DIM + hi * 16;
    const __bf16* bcol1 = bcol0 + (size_t)16 * D_DIM;

    f32x8 acc00 = {}, acc01 = {}, acc10 = {}, acc11 = {};
#pragma unroll 2
    for (int k0 = 0; k0 < D_DIM; k0 += 32) {
        bf16x16 a0 = load_a_16x32(arow0 + k0, hi);
        bf16x16 a1 = load_a_16x32(arow1 + k0, hi);
        bf16x16 b0 = load_b_32x16(bcol0 + k0);
        bf16x16 b1 = load_b_32x16(bcol1 + k0);
        acc00 = wmma_bf16(a0, b0, acc00);
        acc01 = wmma_bf16(a0, b1, acc01);
        acc10 = wmma_bf16(a1, b0, acc10);
        acc11 = wmma_bf16(a1, b1, acc11);
    }

    const f32x8 accs[4] = { acc00, acc01, acc10, acc11 };
#pragma unroll
    for (int ti = 0; ti < 2; ++ti) {
#pragma unroll
        for (int tj = 0; tj < 2; ++tj) {
            const f32x8 acc = accs[ti * 2 + tj];
            const int   n   = n0 + tj * 16 + lo16;
            const float bn  = bias[n];
#pragma unroll
            for (int r = 0; r < 8; ++r) {
                int   m = m0 + ti * 16 + r + hi * 8;
                float y = acc[r] + bn;
                if (which == 0) {
                    Qb[(size_t)m * D_DIM + n] = f2bf(y);
                } else if (which == 1) {
                    Kb[(size_t)m * D_DIM + n] = f2bf(y);
                } else {
                    int bidx = m >> 11;              // m = batch*2048 + s
                    int s    = m & (SEQ - 1);
                    Vt[((size_t)bidx * D_DIM + n) * SEQ + s] = f2bf(y);
                }
            }
        }
    }
}

// ------------------------------------------------------------- scores -------
// S[b][q][k] = (Q[b,q,:] . K[b,k,:]) / sqrt(D), 32x32 tile per wave.

__global__ __launch_bounds__(256) void scores_kernel(
        const __bf16* __restrict__ Qb, const __bf16* __restrict__ Kb,
        float* __restrict__ S) {
    const int batch = blockIdx.z;
    const int wave  = threadIdx.x >> 5;
    const int lane  = threadIdx.x & 31;
    const int lo16  = lane & 15;
    const int hi    = lane >> 4;

    const int gtile = blockIdx.x * 8 + wave;     // 4096 tiles/batch (64 x 64)
    const int q0 = (gtile >> 6) * 32;
    const int k0 = (gtile & 63) * 32;

    const __bf16* qrow0 = Qb + ((size_t)batch * SEQ + q0 + lo16) * D_DIM;
    const __bf16* qrow1 = qrow0 + (size_t)16 * D_DIM;
    const __bf16* kcol0 = Kb + ((size_t)batch * SEQ + k0 + lo16) * D_DIM + hi * 16;
    const __bf16* kcol1 = kcol0 + (size_t)16 * D_DIM;

    f32x8 acc00 = {}, acc01 = {}, acc10 = {}, acc11 = {};
#pragma unroll 2
    for (int d0 = 0; d0 < D_DIM; d0 += 32) {
        __builtin_prefetch(kcol0 + d0 + 128, 0, 0);   // global_prefetch_b8
        bf16x16 a0 = load_a_16x32(qrow0 + d0, hi);
        bf16x16 a1 = load_a_16x32(qrow1 + d0, hi);
        bf16x16 b0 = load_b_32x16(kcol0 + d0);
        bf16x16 b1 = load_b_32x16(kcol1 + d0);
        acc00 = wmma_bf16(a0, b0, acc00);
        acc01 = wmma_bf16(a0, b1, acc01);
        acc10 = wmma_bf16(a1, b0, acc10);
        acc11 = wmma_bf16(a1, b1, acc11);
    }

    const float scale = 0.03125f;                // 1/sqrt(1024)
    float* srow = S + (size_t)batch * SEQ * SEQ;
    const f32x8 accs[4] = { acc00, acc01, acc10, acc11 };
#pragma unroll
    for (int ti = 0; ti < 2; ++ti) {
#pragma unroll
        for (int tj = 0; tj < 2; ++tj) {
            const f32x8 acc = accs[ti * 2 + tj];
#pragma unroll
            for (int r = 0; r < 8; ++r) {
                int m = q0 + ti * 16 + r + hi * 8;
                srow[(size_t)m * SEQ + k0 + tj * 16 + lo16] = acc[r] * scale;
            }
        }
    }
}

// ------------------------------------------------------------- softmax ------
// One block per row; writes bf16 probabilities in place over the fp32 score
// row (bf16 row stride = 2*SEQ elements).  Scores are pre-scaled.

__global__ __launch_bounds__(256) void softmax_kernel(
        float* __restrict__ S, __bf16* __restrict__ Pb) {
    const int row = blockIdx.x;                  // BATCH*SEQ rows
    const int t    = threadIdx.x;
    const int wave = t >> 5;
    const int lane = t & 31;

    const float* rowp = S + (size_t)row * SEQ;
    float x[8];
#pragma unroll
    for (int j = 0; j < 8; ++j) x[j] = rowp[j * 256 + t];

    __shared__ float smem[8];

    float m = x[0];
#pragma unroll
    for (int j = 1; j < 8; ++j) m = fmaxf(m, x[j]);
    for (int off = 16; off >= 1; off >>= 1)
        m = fmaxf(m, __shfl_xor(m, off, 32));
    if (lane == 0) smem[wave] = m;
    __syncthreads();
    float mm = smem[0];
#pragma unroll
    for (int w = 1; w < 8; ++w) mm = fmaxf(mm, smem[w]);
    __syncthreads();

    float e[8], s = 0.f;
#pragma unroll
    for (int j = 0; j < 8; ++j) { e[j] = __expf(x[j] - mm); s += e[j]; }
    for (int off = 16; off >= 1; off >>= 1)
        s += __shfl_xor(s, off, 32);
    if (lane == 0) smem[wave] = s;
    __syncthreads();
    float ss = 0.f;
#pragma unroll
    for (int w = 0; w < 8; ++w) ss += smem[w];
    const float inv = 1.0f / ss;

    __bf16* prow = Pb + (size_t)row * (2 * SEQ);   // aliases S row start
#pragma unroll
    for (int j = 0; j < 8; ++j)
        prow[j * 256 + t] = f2bf(e[j] * inv);
}

// ---------------------------------------------------------------- A @ V -----
// O[b][q][d] = sum_s P[b][q][s] * V[b][s][d], Vt stored [b][d][s].
// 32x32 output tile per wave.

__global__ __launch_bounds__(256) void av_kernel(
        const __bf16* __restrict__ Pb, const __bf16* __restrict__ Vt,
        float* __restrict__ Out) {
    const int batch = blockIdx.z;
    const int wave  = threadIdx.x >> 5;
    const int lane  = threadIdx.x & 31;
    const int lo16  = lane & 15;
    const int hi    = lane >> 4;

    const int gtile = blockIdx.x * 8 + wave;     // 2048 tiles/batch (64 x 32)
    const int q0 = (gtile >> 5) * 32;
    const int d0 = (gtile & 31) * 32;

    const __bf16* prow0 = Pb + ((size_t)batch * SEQ + q0 + lo16) * (2 * SEQ);
    const __bf16* prow1 = prow0 + (size_t)16 * (2 * SEQ);
    const __bf16* vcol0 = Vt + ((size_t)batch * D_DIM + d0 + lo16) * SEQ + hi * 16;
    const __bf16* vcol1 = vcol0 + (size_t)16 * SEQ;

    f32x8 acc00 = {}, acc01 = {}, acc10 = {}, acc11 = {};
#pragma unroll 2
    for (int s0 = 0; s0 < SEQ; s0 += 32) {
        __builtin_prefetch(vcol0 + s0 + 128, 0, 0);
        bf16x16 a0 = load_a_16x32(prow0 + s0, hi);
        bf16x16 a1 = load_a_16x32(prow1 + s0, hi);
        bf16x16 b0 = load_b_32x16(vcol0 + s0);
        bf16x16 b1 = load_b_32x16(vcol1 + s0);
        acc00 = wmma_bf16(a0, b0, acc00);
        acc01 = wmma_bf16(a0, b1, acc01);
        acc10 = wmma_bf16(a1, b0, acc10);
        acc11 = wmma_bf16(a1, b1, acc11);
    }

    float* orow = Out + (size_t)batch * SEQ * D_DIM;
    const f32x8 accs[4] = { acc00, acc01, acc10, acc11 };
#pragma unroll
    for (int ti = 0; ti < 2; ++ti) {
#pragma unroll
        for (int tj = 0; tj < 2; ++tj) {
            const f32x8 acc = accs[ti * 2 + tj];
#pragma unroll
            for (int r = 0; r < 8; ++r) {
                int m = q0 + ti * 16 + r + hi * 8;
                orow[(size_t)m * D_DIM + d0 + tj * 16 + lo16] = acc[r];
            }
        }
    }
}

// ---------------------------------------------------------------- launch ----

extern "C" void kernel_launch(void* const* d_in, const int* in_sizes, int n_in,
                              void* d_out, int out_size, void* d_ws, size_t ws_size,
                              hipStream_t stream) {
    const float* X  = (const float*)d_in[0];
    const float* Wq = (const float*)d_in[1];
    const float* bq = (const float*)d_in[2];
    const float* Wk = (const float*)d_in[3];
    const float* bk = (const float*)d_in[4];
    const float* Wv = (const float*)d_in[5];
    const float* bv = (const float*)d_in[6];
    float* Out = (float*)d_out;

    char* ws = (char*)d_ws;
    const size_t MB = 1024ull * 1024ull;
    __bf16* Xb  = (__bf16*)(ws +   0 * MB);   // 16 MB : [8192][1024] bf16
    __bf16* WtQ = (__bf16*)(ws +  16 * MB);   //  2 MB : [1024][1024] bf16 (T)
    __bf16* WtK = (__bf16*)(ws +  18 * MB);
    __bf16* WtV = (__bf16*)(ws +  20 * MB);
    __bf16* Qb  = (__bf16*)(ws +  22 * MB);   // 16 MB : [8192][1024] bf16
    __bf16* Kb  = (__bf16*)(ws +  38 * MB);
    __bf16* Vt  = (__bf16*)(ws +  54 * MB);   // 16 MB : [4][1024][2048] bf16
    float*  S   = (float*) (ws +  70 * MB);   // 64 MB : [4][2048][2048] f32
    __bf16* Pb  = (__bf16*)(ws +  70 * MB);   // aliases S (bf16 row stride 4096)

    // 1) precision conversion / transposes
    cvt_x_kernel<<<dim3(MROWS * D_DIM / 256), 256, 0, stream>>>(X, Xb);
    cvt_w_kernel<<<dim3(D_DIM * D_DIM / 256, 3), 256, 0, stream>>>(
        Wq, Wk, Wv, WtQ, WtK, WtV);

    // 2) fused Q/K/V projection (8192 32x32 wave-tiles per matrix)
    proj_kernel<<<dim3(1024, 3), 256, 0, stream>>>(
        Xb, WtQ, bq, WtK, bk, WtV, bv, Qb, Kb, Vt);

    // 3) scores = Q.K^T / sqrt(D)  (4096 wave-tiles per batch)
    scores_kernel<<<dim3(512, 1, BATCH), 256, 0, stream>>>(Qb, Kb, S);

    // 4) row-wise softmax -> bf16 probabilities (in place)
    softmax_kernel<<<dim3(MROWS), 256, 0, stream>>>(S, Pb);

    // 5) O = P.V  (2048 wave-tiles per batch)
    av_kernel<<<dim3(256, 1, BATCH), 256, 0, stream>>>(Pb, Vt, Out);
}
